// GAT_10L_54485955117458
// MI455X (gfx1250) — matrix operand order repaired and checked by reading, not compiled, not run on hardware
//
#include <hip/hip_runtime.h>
#include <hip/hip_bf16.h>

// ---------------------------------------------------------------------------
// GATv2 10-layer network for MI455X (gfx1250, wave32, WMMA).
// GEMMs: v_wmma_f32_16x16x32_bf16, register-double-buffered K pipeline,
// 16x(16*NT) strip per wave for A-fragment reuse. Edge softmax/aggregation:
// f32 atomics on an L2-resident working set (192 MB L2 >> 77 MB per layer).
// ---------------------------------------------------------------------------

typedef __attribute__((ext_vector_type(16))) __bf16 v16bf;
typedef __attribute__((ext_vector_type(8)))  float  v8f;

union FragBF { v16bf v; uint4 q[2]; };

#define KDIM 128   // inner dimension of every node GEMM (IN_F == HC == 128)

// ------------------------------ utility kernels ----------------------------

__global__ void k_f32_to_bf16(const float* __restrict__ in,
                              __bf16* __restrict__ out, int n) {
  int i = blockIdx.x * blockDim.x + threadIdx.x;
  if (i < n) out[i] = (__bf16)in[i];
}

// in: f32 [K, Nc] row-major  ->  out: bf16 [Nc, K] (i.e. column of W contiguous)
__global__ void k_transpose_bf16(const float* __restrict__ in,
                                 __bf16* __restrict__ out, int K, int Nc) {
  int i = blockIdx.x * blockDim.x + threadIdx.x;
  if (i >= K * Nc) return;
  int k = i / Nc, n = i % Nc;
  out[n * K + k] = (__bf16)in[i];
}

__global__ void k_fill(float* __restrict__ p, float v, int n) {
  int i = blockIdx.x * blockDim.x + threadIdx.x;
  if (i < n) p[i] = v;
}

// ------------------------------ WMMA GEMM ----------------------------------
// C[M, Ncols] = A[M,128](bf16 row-major) x B[128,Ncols] given as BT[Ncols,128].
// One wave computes a 16 x (16*NT) strip: A fragment reused NT times, NT
// back-to-back WMMAs per K-step, ping/pong register buffers so next K-block
// loads overlap current WMMAs.
// Fragment layouts per CDNA5 ISA 7.12.2:
//   A (16x32 bf16): half h = lane>>4, row = lane&15,
//       VGPR0..3 = K[h*8 .. h*8+7], VGPR4..7 = K[16+h*8 .. 16+h*8+7]
//   B (32x16 bf16): col = lane&15, same K order as A
//   D (16x16 f32): col = lane&15, VGPR j -> row j + h*8
template <int NT>
__global__ void __launch_bounds__(128)
k_gemm_wmma_bf16(const __bf16* __restrict__ A,
                 const __bf16* __restrict__ BT,
                 float* __restrict__ C,
                 int Ncols, int totalTiles) {
  int wid  = threadIdx.x >> 5;
  int lane = threadIdx.x & 31;
  int tile = blockIdx.x * 4 + wid;
  if (tile >= totalTiles) return;
  int ntGroups = Ncols / (16 * NT);
  int mtile = tile / ntGroups;
  int group = tile % ntGroups;
  int l15   = lane & 15;
  int khalf = lane >> 4;

  const __bf16* arow = A + (size_t)((mtile << 4) + l15) * KDIM;
  const __bf16* bcol[NT];
#pragma unroll
  for (int j = 0; j < NT; ++j)
    bcol[j] = BT + (size_t)(((group * NT + j) << 4) + l15) * KDIM;

  FragBF aP, aQ;
  FragBF bP[NT], bQ[NT];
  v8f acc[NT];
#pragma unroll
  for (int j = 0; j < NT; ++j) acc[j] = (v8f){};

  auto loadA = [&](FragBF& f, int kb) {
    f.q[0] = *(const uint4*)(arow + kb + khalf * 8);
    f.q[1] = *(const uint4*)(arow + kb + 16 + khalf * 8);
  };
  auto loadB = [&](FragBF (&f)[NT], int kb) {
#pragma unroll
    for (int j = 0; j < NT; ++j) {
      const uint4* pb = (const uint4*)(bcol[j] + kb + khalf * 8);
      f[j].q[0] = pb[0];
      f[j].q[1] = pb[2];
    }
  };
  auto mm = [&](FragBF& a, FragBF (&b)[NT]) {
#pragma unroll
    for (int j = 0; j < NT; ++j)
      acc[j] = __builtin_amdgcn_wmma_f32_16x16x32_bf16(
          false, a.v, false, b[j].v, (short)0, acc[j], false, false);
  };

  // K = 128 = 4 steps of 32, ping/pong double buffered
  loadA(aP, 0);   loadB(bP, 0);
  loadA(aQ, 32);  loadB(bQ, 32);
  mm(aP, bP);
  loadA(aP, 64);  loadB(bP, 64);
  mm(aQ, bQ);
  loadA(aQ, 96);  loadB(bQ, 96);
  mm(aP, bP);
  mm(aQ, bQ);

  // store D tiles
#pragma unroll
  for (int j = 0; j < NT; ++j) {
    float* cb = C + (size_t)((mtile << 4) + (khalf << 3)) * Ncols
                  + (((group * NT + j) << 4) + l15);
#pragma unroll
    for (int r = 0; r < 8; ++r)
      cb[(size_t)r * Ncols] = acc[j][r];
  }
}

// ------------------------------ edge kernels -------------------------------

__device__ inline void atomicMaxF(float* addr, float val) {
  // sign-split trick: valid IEEE float max via integer atomics
  if (val >= 0.0f) atomicMax((int*)addr, __float_as_int(val));
  else             atomicMin((unsigned int*)addr, __float_as_uint(val));
}

// Wave per edge: e[e,h] = sum_c leakyrelu(xl[s]+xr[d]+ea*We, 0.2) * att[h,c]
// plus running per-(dst,head) max for the segment softmax.
template <int H, int C>
__global__ void __launch_bounds__(256)
k_edge_logits(const float* __restrict__ XL, const float* __restrict__ XR,
              const float* __restrict__ ea, const float* __restrict__ We,
              const float* __restrict__ att,
              const int* __restrict__ src, const int* __restrict__ dst,
              float* __restrict__ elog, float* __restrict__ nmax, int nE) {
  const int HC = H * C;
  const int lanesPerHead = 32 / H;
  const int cpl = C / lanesPerHead;           // channels per lane
  int lane = threadIdx.x & 31;
  int e = (blockIdx.x * blockDim.x + threadIdx.x) >> 5;
  if (e >= nE) return;
  int s = src[e], d = dst[e];
  float eav = ea[e];
  int head   = lane / lanesPerHead;
  int within = lane % lanesPerHead;
  int ch0    = head * C + within * cpl;
  const float* xl = XL + (size_t)s * HC;
  const float* xr = XR + (size_t)d * HC;
  float partial = 0.f;
#pragma unroll
  for (int t = 0; t < cpl; ++t) {
    int ch = ch0 + t;
    float m  = xl[ch] + xr[ch] + eav * We[ch];
    float lr = m > 0.f ? m : 0.2f * m;
    partial += lr * att[ch];
  }
#pragma unroll
  for (int off = lanesPerHead >> 1; off > 0; off >>= 1)
    partial += __shfl_xor(partial, off, 32);
  if (within == 0) {
    elog[(size_t)e * H + head] = partial;
    atomicMaxF(&nmax[(size_t)d * H + head], partial);
  }
}

__global__ void k_edge_softmax(float* __restrict__ elog,
                               const float* __restrict__ nmax,
                               float* __restrict__ denom,
                               const int* __restrict__ dst, int nE, int H_) {
  int i = blockIdx.x * blockDim.x + threadIdx.x;
  if (i >= nE * H_) return;
  int e = i / H_, h = i % H_;
  int d = dst[e];
  float v = __expf(elog[i] - nmax[(size_t)d * H_ + h]);
  elog[i] = v;
  atomicAdd(&denom[(size_t)d * H_ + h], v);
}

// Wave per edge: agg[d] += alpha[e,h] * xl[s]
template <int H, int C>
__global__ void __launch_bounds__(256)
k_edge_aggregate(const float* __restrict__ XL, const float* __restrict__ elog,
                 const float* __restrict__ denom,
                 const int* __restrict__ src, const int* __restrict__ dst,
                 float* __restrict__ agg, int nE) {
  const int HC = H * C;
  const int cpl = HC / 32;
  int lane = threadIdx.x & 31;
  int e = (blockIdx.x * blockDim.x + threadIdx.x) >> 5;
  if (e >= nE) return;
  int s = src[e], d = dst[e];
  int ch0 = lane * cpl;
  int head = ch0 / C;   // cpl channels stay within one head
  float alpha = elog[(size_t)e * H + head] /
                (denom[(size_t)d * H + head] + 1e-16f);
  const float* xl = XL + (size_t)s * HC;
  float* ag = agg + (size_t)d * HC;
#pragma unroll
  for (int t = 0; t < cpl; ++t) {
    int ch = ch0 + t;
    atomicAdd(&ag[ch], alpha * xl[ch]);
  }
}

// h = bn(relu(agg + bias)); emit bf16 activations for the next layer's GEMM
__global__ void k_bias_relu_bn(const float* __restrict__ agg,
                               const float* __restrict__ bias,
                               const float* __restrict__ g,
                               const float* __restrict__ b,
                               const float* __restrict__ m,
                               const float* __restrict__ v,
                               __bf16* __restrict__ actbf, int n, int HC_) {
  int i = blockIdx.x * blockDim.x + threadIdx.x;
  if (i >= n) return;
  int j = i % HC_;
  float h = agg[i] + bias[j];
  h = h > 0.f ? h : 0.f;
  h = (h - m[j]) * rsqrtf(v[j] + 1e-5f) * g[j] + b[j];
  actbf[i] = (__bf16)h;
}

// Layer 10 tail: bn10(relu(agg+b10)) @ lin_W + lin_b  ->  out [N,4]
__global__ void k_final(const float* __restrict__ agg,      // [N,32]
                        const float* __restrict__ b10,
                        const float* __restrict__ g, const float* __restrict__ b,
                        const float* __restrict__ m, const float* __restrict__ v,
                        const float* __restrict__ linW,     // [32,4]
                        const float* __restrict__ linb,     // [4]
                        float* __restrict__ out, int Nn) {
  int n = blockIdx.x * blockDim.x + threadIdx.x;
  if (n >= Nn) return;
  float o0 = linb[0], o1 = linb[1], o2 = linb[2], o3 = linb[3];
  const float* a = agg + (size_t)n * 32;
#pragma unroll
  for (int c = 0; c < 32; ++c) {
    float h = a[c] + b10[c];
    h = h > 0.f ? h : 0.f;
    h = (h - m[c]) * rsqrtf(v[c] + 1e-5f) * g[c] + b[c];
    o0 += h * linW[c * 4 + 0];
    o1 += h * linW[c * 4 + 1];
    o2 += h * linW[c * 4 + 2];
    o3 += h * linW[c * 4 + 3];
  }
  float* op = out + (size_t)n * 4;
  op[0] = o0; op[1] = o1; op[2] = o2; op[3] = o3;
}

// ------------------------------ host driver --------------------------------

extern "C" void kernel_launch(void* const* d_in, const int* in_sizes, int n_in,
                              void* d_out, int out_size, void* d_ws, size_t ws_size,
                              hipStream_t stream) {
  (void)in_sizes; (void)n_in; (void)out_size; (void)ws_size;
  const int N_ = 50000, E_ = 800000, HC_ = 128, C_ = 32, H_ = 4;

  const float* x      = (const float*)d_in[0];
  const int*   ei     = (const int*)  d_in[1];
  const float* ea     = (const float*)d_in[2];
  const float* l1_Wl  = (const float*)d_in[3];
  const float* l1_Wr  = (const float*)d_in[4];
  const float* l1_We  = (const float*)d_in[5];
  const float* l1_att = (const float*)d_in[6];
  const float* l1_b   = (const float*)d_in[7];
  const float* mid_Wl = (const float*)d_in[8];
  const float* mid_Wr = (const float*)d_in[9];
  const float* mid_We = (const float*)d_in[10];
  const float* mid_att= (const float*)d_in[11];
  const float* mid_b  = (const float*)d_in[12];
  const float* l10_Wl = (const float*)d_in[13];
  const float* l10_Wr = (const float*)d_in[14];
  const float* l10_We = (const float*)d_in[15];
  const float* l10_att= (const float*)d_in[16];
  const float* l10_b  = (const float*)d_in[17];
  const float* bn_g   = (const float*)d_in[18];
  const float* bn_b   = (const float*)d_in[19];
  const float* bn_m   = (const float*)d_in[20];
  const float* bn_v   = (const float*)d_in[21];
  const float* bn10_g = (const float*)d_in[22];
  const float* bn10_b = (const float*)d_in[23];
  const float* bn10_m = (const float*)d_in[24];
  const float* bn10_v = (const float*)d_in[25];
  const float* lin_W  = (const float*)d_in[26];
  const float* lin_b  = (const float*)d_in[27];
  const int* src = ei;
  const int* dst = ei + E_;
  float* out = (float*)d_out;

  // workspace carve-out (256B aligned slices)
  char* wp = (char*)d_ws;
  auto alloc = [&](size_t bytes) -> char* {
    char* p = wp; wp += (bytes + 255) & ~(size_t)255; return p;
  };
  __bf16* actbf = (__bf16*)alloc((size_t)N_ * HC_ * 2);
  float*  XL    = (float*) alloc((size_t)N_ * HC_ * 4);
  float*  XR    = (float*) alloc((size_t)N_ * HC_ * 4);
  float*  agg   = (float*) alloc((size_t)N_ * HC_ * 4);
  float*  elog  = (float*) alloc((size_t)E_ * H_  * 4);
  float*  nmax  = (float*) alloc((size_t)N_ * H_  * 4);
  float*  den   = (float*) alloc((size_t)N_ * H_  * 4);
  __bf16* WlT   = (__bf16*)alloc((size_t)9 * HC_ * HC_ * 2);  // layers 1..9
  __bf16* WrT   = (__bf16*)alloc((size_t)9 * HC_ * HC_ * 2);
  __bf16* Wl10T = (__bf16*)alloc((size_t)C_ * HC_ * 2);
  __bf16* Wr10T = (__bf16*)alloc((size_t)C_ * HC_ * 2);

  const int TB = 256;
  auto cdiv = [](long long a, long long b) { return (int)((a + b - 1) / b); };

  // ---- weight prep: transpose + bf16 convert (per call; no cached state) ----
  {
    int g128 = cdiv((long long)HC_ * HC_, TB);
    k_transpose_bf16<<<g128, TB, 0, stream>>>(l1_Wl, WlT, HC_, HC_);
    k_transpose_bf16<<<g128, TB, 0, stream>>>(l1_Wr, WrT, HC_, HC_);
    for (int i = 0; i < 8; ++i) {
      k_transpose_bf16<<<g128, TB, 0, stream>>>(mid_Wl + (size_t)i * HC_ * HC_,
                                                WlT + (size_t)(i + 1) * HC_ * HC_, HC_, HC_);
      k_transpose_bf16<<<g128, TB, 0, stream>>>(mid_Wr + (size_t)i * HC_ * HC_,
                                                WrT + (size_t)(i + 1) * HC_ * HC_, HC_, HC_);
    }
    int g32 = cdiv((long long)HC_ * C_, TB);
    k_transpose_bf16<<<g32, TB, 0, stream>>>(l10_Wl, Wl10T, HC_, C_);
    k_transpose_bf16<<<g32, TB, 0, stream>>>(l10_Wr, Wr10T, HC_, C_);
  }

  // strip-tile counts: NT=4 (128 cols -> 2 groups), NT=2 (32 cols -> 1 group)
  const int tiles128 = (N_ / 16) * (HC_ / 64);   // 6250
  const int tiles32  = (N_ / 16) * (C_  / 32);   // 3125
  const int gEdgeWave = cdiv((long long)E_ * 32, TB);
  const float NEG_INF = -__builtin_inff();

  // first activations: x -> bf16
  k_f32_to_bf16<<<cdiv((long long)N_ * HC_, TB), TB, 0, stream>>>(x, actbf, N_ * HC_);

  // ---- layers 1..9 (H=4, C=32, HC=128) ----
  for (int L = 0; L < 9; ++L) {
    const float* We_l  = (L == 0) ? l1_We  : mid_We  + (size_t)(L - 1) * HC_;
    const float* att_l = (L == 0) ? l1_att : mid_att + (size_t)(L - 1) * H_ * C_;
    const float* b_l   = (L == 0) ? l1_b   : mid_b   + (size_t)(L - 1) * HC_;
    const __bf16* wl = WlT + (size_t)L * HC_ * HC_;
    const __bf16* wr = WrT + (size_t)L * HC_ * HC_;

    k_gemm_wmma_bf16<4><<<cdiv(tiles128, 4), 128, 0, stream>>>(actbf, wl, XL, HC_, tiles128);
    k_gemm_wmma_bf16<4><<<cdiv(tiles128, 4), 128, 0, stream>>>(actbf, wr, XR, HC_, tiles128);

    k_fill<<<cdiv((long long)N_ * H_, TB), TB, 0, stream>>>(nmax, NEG_INF, N_ * H_);
    k_fill<<<cdiv((long long)N_ * H_, TB), TB, 0, stream>>>(den, 0.f, N_ * H_);
    k_fill<<<cdiv((long long)N_ * HC_, TB), TB, 0, stream>>>(agg, 0.f, N_ * HC_);

    k_edge_logits<4, 32><<<gEdgeWave, TB, 0, stream>>>(XL, XR, ea, We_l, att_l,
                                                       src, dst, elog, nmax, E_);
    k_edge_softmax<<<cdiv((long long)E_ * H_, TB), TB, 0, stream>>>(elog, nmax, den,
                                                                    dst, E_, H_);
    k_edge_aggregate<4, 32><<<gEdgeWave, TB, 0, stream>>>(XL, elog, den, src, dst,
                                                          agg, E_);
    k_bias_relu_bn<<<cdiv((long long)N_ * HC_, TB), TB, 0, stream>>>(
        agg, b_l, bn_g + (size_t)L * HC_, bn_b + (size_t)L * HC_,
        bn_m + (size_t)L * HC_, bn_v + (size_t)L * HC_, actbf, N_ * HC_, HC_);
  }

  // ---- layer 10 (H=1, C=32, concat=False -> mean over 1 head == identity) ----
  k_gemm_wmma_bf16<2><<<cdiv(tiles32, 4), 128, 0, stream>>>(actbf, Wl10T, XL, C_, tiles32);
  k_gemm_wmma_bf16<2><<<cdiv(tiles32, 4), 128, 0, stream>>>(actbf, Wr10T, XR, C_, tiles32);

  k_fill<<<cdiv((long long)N_, TB), TB, 0, stream>>>(nmax, NEG_INF, N_);
  k_fill<<<cdiv((long long)N_, TB), TB, 0, stream>>>(den, 0.f, N_);
  k_fill<<<cdiv((long long)N_ * C_, TB), TB, 0, stream>>>(agg, 0.f, N_ * C_);

  k_edge_logits<1, 32><<<gEdgeWave, TB, 0, stream>>>(XL, XR, ea, l10_We, l10_att,
                                                     src, dst, elog, nmax, E_);
  k_edge_softmax<<<cdiv((long long)E_, TB), TB, 0, stream>>>(elog, nmax, den, dst, E_, 1);
  k_edge_aggregate<1, 32><<<gEdgeWave, TB, 0, stream>>>(XL, elog, den, src, dst, agg, E_);

  k_final<<<cdiv((long long)N_, TB), TB, 0, stream>>>(agg, l10_b, bn10_g, bn10_b,
                                                      bn10_m, bn10_v, lin_W, lin_b,
                                                      out, N_);
}
